// GRUAnomalyDetector_36790689858267
// MI455X (gfx1250) — matrix operand order repaired and checked by reading, not compiled, not run on hardware
//
#include <hip/hip_runtime.h>

typedef __bf16 bf16;
typedef bf16  v16bf __attribute__((ext_vector_type(16)));
typedef float v8f   __attribute__((ext_vector_type(8)));
typedef int   v4i   __attribute__((ext_vector_type(4)));

static constexpr int kB = 256, kT = 2048, kIN = 16, kH = 128, kG = 384;

union Frag {
    v16bf v;
    uint4 q[2];
    bf16  h[16];
};
union Half8 {          // 8 bf16 = one B128
    uint4 q;
    bf16  h[8];
};

__device__ __forceinline__ float sigm(float x) { return 1.f / (1.f + __expf(-x)); }
__device__ __forceinline__ float tanh_fast(float x) { return 1.f - 2.f / (__expf(2.f * x) + 1.f); }

// ---- CDNA5 async copy global->LDS (ASYNCcnt), with portable fallback -------
__device__ __forceinline__ void async_copy16(const bf16* __restrict__ g, bf16* l) {
#if __has_builtin(__builtin_amdgcn_global_load_async_to_lds_b128)
    __builtin_amdgcn_global_load_async_to_lds_b128(
        (__attribute__((address_space(1))) v4i*)g,
        (__attribute__((address_space(3))) v4i*)l, 0, 0);
#else
    *(uint4*)l = *(const uint4*)g;
#endif
}

template <int N>
__device__ __forceinline__ void wait_asynccnt() {
#if __has_builtin(__builtin_amdgcn_s_wait_asynccnt)
    __builtin_amdgcn_s_wait_asynccnt((unsigned short)N);
#else
    asm volatile("s_wait_asynccnt %0" ::"i"(N) : "memory");
#endif
}

// ---------------------------------------------------------------------------
// K0: repack weights (f32, row-major (N,K)) into WMMA B-fragment order (bf16).
// Fragment (jt,kk) holds B[k][n] = W[n][k]; lane L<16 holds k-locals 0..7,16..23,
// lane>=16 holds 8..15,24..31; column n = jt*16+(L&15).  512 bf16 per fragment.
// Fragments in order: Whh0[96] Whh1[96] Wih1[96] Wih0[24] W1[16]  (328 total)
// ---------------------------------------------------------------------------
__global__ __launch_bounds__(256) void pack_weights(
    const float* __restrict__ Whh0, const float* __restrict__ Whh1,
    const float* __restrict__ Wih1, const float* __restrict__ Wih0,
    const float* __restrict__ W1, bf16* __restrict__ pack) {
    int f = blockIdx.x;
    const float* W;
    int Kreal, Kst, jt, kk;
    if (f < 96)        { W = Whh0; Kreal = 128; Kst = 128; int fl = f;       jt = fl >> 2; kk = fl & 3; }
    else if (f < 192)  { W = Whh1; Kreal = 128; Kst = 128; int fl = f - 96;  jt = fl >> 2; kk = fl & 3; }
    else if (f < 288)  { W = Wih1; Kreal = 128; Kst = 128; int fl = f - 192; jt = fl >> 2; kk = fl & 3; }
    else if (f < 312)  { W = Wih0; Kreal = 16;  Kst = 16;  jt = f - 288;     kk = 0; }
    else               { W = W1;   Kreal = 128; Kst = 128; int fl = f - 312; jt = fl >> 2; kk = fl & 3; }
    for (int e = threadIdx.x; e < 512; e += 256) {
        int lane = e >> 4, s = e & 15;
        int n = jt * 16 + (lane & 15);
        int k = kk * 32 + ((s < 8) ? ((lane >> 4) * 8 + s) : (16 + (lane >> 4) * 8 + (s - 8)));
        float v = (k < Kreal) ? W[n * Kst + k] : 0.f;
        pack[(size_t)f * 512 + e] = (bf16)v;
    }
}

// ---------------------------------------------------------------------------
// K1: gi0[t][g][b] = x[b][t][:16] . Wih0[g][:16] + bih0[g]   (bf16, (T,3H,B))
// One block = one (t, 16-batch) tile; 8 waves x 3 g-tiles x 1 WMMA (K=32 pad).
// ---------------------------------------------------------------------------
__global__ __launch_bounds__(256) void gi0_kernel(
    const float* __restrict__ x, const bf16* __restrict__ pWih0,
    const float* __restrict__ bih0, bf16* __restrict__ gi) {
    int bid = blockIdx.x;
    int t = bid >> 4;
    int b0 = (bid & 15) * 16;
    int tid = threadIdx.x, w = tid >> 5, lane = tid & 31;

    // A fragment: 16x32 bf16, rows = batch, K 0..15 = x, 16..31 = 0
    int row = b0 + (lane & 15);
    const float* xs = x + ((size_t)row * kT + t) * kIN + ((lane < 16) ? 0 : 8);
    float xv[8];
    *(uint4*)&xv[0] = *(const uint4*)xs;
    *(uint4*)&xv[4] = *(const uint4*)(xs + 4);
    Frag a;
    for (int i = 0; i < 8; ++i) a.h[i] = (bf16)xv[i];
    for (int i = 8; i < 16; ++i) a.h[i] = (bf16)0.f;

    int mb = (lane < 16) ? 0 : 8;
    for (int g3 = 0; g3 < 3; ++g3) {
        int jt = g3 * 8 + w;
        Frag b;
        const uint4* bp = (const uint4*)(pWih0 + (size_t)jt * 512 + lane * 16);
        b.q[0] = bp[0]; b.q[1] = bp[1];
        v8f acc = {0.f, 0.f, 0.f, 0.f, 0.f, 0.f, 0.f, 0.f};
        acc = __builtin_amdgcn_wmma_f32_16x16x32_bf16(false, a.v, false, b.v,
                                                      (short)0, acc, false, false);
        int col = jt * 16 + (lane & 15);
        float bias = bih0[col];
        Half8 st;
        for (int vv = 0; vv < 8; ++vv) st.h[vv] = (bf16)(acc[vv] + bias);
        // (T, G, B): lane's 8 rows are consecutive -> one B128 store
        *(uint4*)(gi + ((size_t)t * kG + col) * kB + b0 + mb) = st.q;
    }
}

// ---------------------------------------------------------------------------
// K2/K4: sequential GRU scan over T.  One block per 16-row batch tile.
// Wave w owns column tile w of H for all three gates (r/z/n lane-aligned).
// Whh B-fragments + previous h (per-lane) held in registers for the whole
// scan; gi_t AND h double-buffered in LDS -> exactly ONE barrier per step.
// gi layout (T, 3H, B): LDS tile is [g][b] so each gate read is one B128.
// ---------------------------------------------------------------------------
__global__ __launch_bounds__(256) void gru_scan_kernel(
    const bf16* __restrict__ gi,    // (T, 3H, B) bf16, bih folded in
    const bf16* __restrict__ pWhh,  // 96 packed fragments
    const float* __restrict__ bhh,  // (3H) f32
    bf16* __restrict__ hout) {      // (T, B, H) bf16
    __shared__ __align__(16) bf16 s_gi[2][kG * 16];  // 2 x 12 KB, [g][b]
    __shared__ __align__(16) bf16 s_h[2][16 * kH];   // 2 x 4 KB, [b][k]
    __shared__ float s_bhh[kG];

    int tid = threadIdx.x, w = tid >> 5, lane = tid & 31;
    int b0 = blockIdx.x * 16;

    // persistent recurrent-weight fragments (12 x 8 VGPRs)
    Frag bw[3][4];
    for (int g3 = 0; g3 < 3; ++g3)
        for (int kk = 0; kk < 4; ++kk) {
            int jt = g3 * 8 + w;
            const uint4* bp = (const uint4*)(pWhh + (size_t)(jt * 4 + kk) * 512 + lane * 16);
            bw[g3][kk].q[0] = bp[0];
            bw[g3][kk].q[1] = bp[1];
        }
    if (tid < kG) s_bhh[tid] = bhh[tid];
    if (tid < kG - 256) s_bhh[tid + 256] = bhh[tid + 256];
    {   // h0 = 0 in buffer 0
        uint4 z = {0, 0, 0, 0};
        ((uint4*)s_h[0])[tid] = z;
    }

    // prime buffer 0 with gi_0: chunk c covers [g = c/2][half = c&1]
    {
        const bf16* g0 = gi + (size_t)b0;
        for (int i = 0; i < 3; ++i) {
            int c = tid + i * 256;
            async_copy16(g0 + (c >> 1) * kB + (c & 1) * 8, &s_gi[0][c * 8]);
        }
    }

    int col = lane & 15;
    int mb = (lane < 16) ? 0 : 8;
    int koff = (lane < 16) ? 0 : 8;
    int r0 = lane & 15;
    int colg[3];
    v8f cinit[3];
    for (int g3 = 0; g3 < 3; ++g3) colg[g3] = g3 * 128 + w * 16 + col;
    float hprev[8];
    for (int vv = 0; vv < 8; ++vv) hprev[vv] = 0.f;

    wait_asynccnt<0>();
    __syncthreads();   // publish gi_0 + zeroed h buffer
    for (int g3 = 0; g3 < 3; ++g3) {
        float bv = s_bhh[colg[g3]];
        for (int vv = 0; vv < 8; ++vv) cinit[g3][vv] = bv;  // bhh folded into C
    }

    for (int t = 0; t < kT; ++t) {
        int p = t & 1;
        // coalesced store of h_{t-1} (stable in s_h[p] during this step)
        if (t > 0)
            ((uint4*)(hout + ((size_t)(t - 1) * kB + b0) * kH))[tid] =
                ((const uint4*)s_h[p])[tid];
        if (t + 1 < kT) {   // issue async fetch of gi_{t+1} into other buffer
            const bf16* gn = gi + (size_t)(t + 1) * kG * kB + b0;
            bf16* dn = s_gi[1 - p];
            for (int i = 0; i < 3; ++i) {
                int c = tid + i * 256;
                async_copy16(gn + (c >> 1) * kB + (c & 1) * 8, dn + c * 8);
            }
        }

        // A fragments from current h (shared by all 3 gate tiles)
        Frag a[4];
        for (int kk = 0; kk < 4; ++kk) {
            const bf16* ap = s_h[p] + r0 * kH + kk * 32 + koff;
            a[kk].q[0] = *(const uint4*)ap;
            a[kk].q[1] = *(const uint4*)(ap + 16);
        }
        v8f acc[3];
        for (int g3 = 0; g3 < 3; ++g3) {
            v8f c = cinit[g3];
            for (int kk = 0; kk < 4; ++kk)
                c = __builtin_amdgcn_wmma_f32_16x16x32_bf16(false, a[kk].v, false,
                                                            bw[g3][kk].v, (short)0,
                                                            c, false, false);
            acc[g3] = c;
        }

        // per-gate gi values: 8 consecutive bf16 each (transposed LDS tile)
        const bf16* sg = s_gi[p];
        Half8 gv[3];
        for (int g3 = 0; g3 < 3; ++g3)
            gv[g3].q = *(const uint4*)(sg + colg[g3] * 16 + mb);

        // gates + state update (f32 math); h_old lives in registers
        float hnew[8];
        for (int vv = 0; vv < 8; ++vv) {
            float rr = sigm((float)gv[0].h[vv] + acc[0][vv]);
            float zz = sigm((float)gv[1].h[vv] + acc[1][vv]);
            float nn = tanh_fast((float)gv[2].h[vv] + rr * acc[2][vv]);
            hnew[vv] = (1.f - zz) * nn + zz * hprev[vv];
            hprev[vv] = hnew[vv];
        }
        // write h_t into the other h buffer (no conflict with s_h[p] readers)
        for (int vv = 0; vv < 8; ++vv)
            s_h[1 - p][(mb + vv) * kH + w * 16 + col] = (bf16)hnew[vv];

        if (t + 1 < kT) wait_asynccnt<3>();  // own gi_{t+1} copies landed
        else            wait_asynccnt<0>();
        __syncthreads();   // single barrier: publishes s_h[1-p] AND gi_{t+1}
    }
    // final h_{T-1} lives in s_h[kT & 1]
    ((uint4*)(hout + ((size_t)(kT - 1) * kB + b0) * kH))[tid] =
        ((const uint4*)s_h[kT & 1])[tid];
}

// ---------------------------------------------------------------------------
// K3: gi1[t][g][b] = h1 @ Wih1^T + bih1, fully time-parallel. 16 rows/block.
// ---------------------------------------------------------------------------
__global__ __launch_bounds__(256) void proj_kernel(
    const bf16* __restrict__ hin,    // (T*B, H) bf16
    const bf16* __restrict__ pW,     // 96 packed fragments
    const float* __restrict__ bias,  // (3H)
    bf16* __restrict__ gi) {         // (T, 3H, B) bf16
    int row0 = blockIdx.x * 16;      // flat row = t*B + b; 16 rows share t-page
    int tid = threadIdx.x, w = tid >> 5, lane = tid & 31;
    int r = row0 + (lane & 15);
    int koff = (lane < 16) ? 0 : 8;

    Frag a[4];
    for (int kk = 0; kk < 4; ++kk) {
        const bf16* ap = hin + (size_t)r * kH + kk * 32 + koff;
        a[kk].q[0] = *(const uint4*)ap;
        a[kk].q[1] = *(const uint4*)(ap + 16);
    }
    int t = row0 >> 8;       // B = 256
    int b0 = row0 & 255;
    int mb = (lane < 16) ? 0 : 8;
    for (int g3 = 0; g3 < 3; ++g3) {
        int jt = g3 * 8 + w;
        v8f acc = {0.f, 0.f, 0.f, 0.f, 0.f, 0.f, 0.f, 0.f};
        for (int kk = 0; kk < 4; ++kk) {
            Frag b;
            const uint4* bp = (const uint4*)(pW + (size_t)(jt * 4 + kk) * 512 + lane * 16);
            b.q[0] = bp[0]; b.q[1] = bp[1];
            acc = __builtin_amdgcn_wmma_f32_16x16x32_bf16(false, a[kk].v, false, b.v,
                                                          (short)0, acc, false, false);
        }
        int col = jt * 16 + (lane & 15);
        float bv = bias[col];
        Half8 st;
        for (int vv = 0; vv < 8; ++vv) st.h[vv] = (bf16)(acc[vv] + bv);
        *(uint4*)(gi + ((size_t)t * kG + col) * kB + b0 + mb) = st.q;
    }
}

// ---------------------------------------------------------------------------
// K5: scores = relu(h2 @ W1^T + b1) . W2 + b2   -> out (B, T) f32
// One block = 16 flat rows (fixed t, 16 b's). 4 waves, one 16-col tile each.
// ---------------------------------------------------------------------------
__global__ __launch_bounds__(128) void head_kernel(
    const bf16* __restrict__ h2,   // (T*B, H) bf16
    const bf16* __restrict__ pW1,  // 16 packed fragments
    const float* __restrict__ b1, const float* __restrict__ W2,
    const float* __restrict__ b2, float* __restrict__ out) {
    __shared__ float s_acc[16];
    int row0 = blockIdx.x * 16;
    int t = row0 >> 8;       // flat row = t*B + b, B = 256
    int b0 = row0 & 255;
    int tid = threadIdx.x, w = tid >> 5, lane = tid & 31;
    int r = row0 + (lane & 15);
    int koff = (lane < 16) ? 0 : 8;

    Frag a[4];
    for (int kk = 0; kk < 4; ++kk) {
        const bf16* ap = h2 + (size_t)r * kH + kk * 32 + koff;
        a[kk].q[0] = *(const uint4*)ap;
        a[kk].q[1] = *(const uint4*)(ap + 16);
    }
    v8f acc = {0.f, 0.f, 0.f, 0.f, 0.f, 0.f, 0.f, 0.f};
    for (int kk = 0; kk < 4; ++kk) {
        Frag b;
        const uint4* bp = (const uint4*)(pW1 + (size_t)(w * 4 + kk) * 512 + lane * 16);
        b.q[0] = bp[0]; b.q[1] = bp[1];
        acc = __builtin_amdgcn_wmma_f32_16x16x32_bf16(false, a[kk].v, false, b.v,
                                                      (short)0, acc, false, false);
    }
    if (tid < 16) s_acc[tid] = 0.f;
    __syncthreads();
    int col = w * 16 + (lane & 15);
    float bcol = b1[col], wcol = W2[col];
    int mb = (lane < 16) ? 0 : 8;
    for (int vv = 0; vv < 8; ++vv) {
        float hid = acc[vv] + bcol;
        hid = hid > 0.f ? hid : 0.f;
        atomicAdd(&s_acc[mb + vv], hid * wcol);   // ds_add_f32
    }
    __syncthreads();
    if (tid < 16)
        out[(size_t)(b0 + tid) * kT + t] = s_acc[tid] + b2[0];
}

// ---------------------------------------------------------------------------
extern "C" void kernel_launch(void* const* d_in, const int* in_sizes, int n_in,
                              void* d_out, int out_size, void* d_ws, size_t ws_size,
                              hipStream_t stream) {
    (void)in_sizes; (void)n_in; (void)out_size; (void)ws_size;
    const float* x    = (const float*)d_in[0];
    const float* Wih0 = (const float*)d_in[1];
    const float* Whh0 = (const float*)d_in[2];
    const float* bih0 = (const float*)d_in[3];
    const float* bhh0 = (const float*)d_in[4];
    const float* Wih1 = (const float*)d_in[5];
    const float* Whh1 = (const float*)d_in[6];
    const float* bih1 = (const float*)d_in[7];
    const float* bhh1 = (const float*)d_in[8];
    const float* W1   = (const float*)d_in[9];
    const float* b1   = (const float*)d_in[10];
    const float* W2   = (const float*)d_in[11];
    const float* b2   = (const float*)d_in[12];
    float* out = (float*)d_out;

    // workspace layout
    bf16* pack = (bf16*)d_ws;                                   // 328*512 bf16
    bf16* gi   = (bf16*)((char*)d_ws + (size_t)512 * 1024);
    bf16* hbuf = (bf16*)((char*)d_ws + (size_t)512 * 1024 + (size_t)kT * kB * kG * 2);
    bf16* pWhh0 = pack;
    bf16* pWhh1 = pack + 96 * 512;
    bf16* pWih1 = pack + 192 * 512;
    bf16* pWih0 = pack + 288 * 512;
    bf16* pW1   = pack + 312 * 512;

    pack_weights<<<328, 256, 0, stream>>>(Whh0, Whh1, Wih1, Wih0, W1, pack);
    gi0_kernel<<<kT * (kB / 16), 256, 0, stream>>>(x, pWih0, bih0, gi);
    gru_scan_kernel<<<kB / 16, 256, 0, stream>>>(gi, pWhh0, bhh0, hbuf);   // layer 0
    proj_kernel<<<(kT * kB) / 16, 256, 0, stream>>>(hbuf, pWih1, bih1, gi);
    gru_scan_kernel<<<kB / 16, 256, 0, stream>>>(gi, pWhh1, bhh1, hbuf);   // layer 1
    head_kernel<<<(kT * kB) / 16, 128, 0, stream>>>(hbuf, pW1, b1, W2, b2, out);
}
